// CorefPairScorer_7997229105345
// MI455X (gfx1250) — matrix operand order repaired and checked by reading, not compiled, not run on hardware
//
#include <hip/hip_runtime.h>
#include <hip/hip_bf16.h>
#include <math.h>

// ---------------- problem constants ----------------
#define B_    4
#define T_    4096
#define D_    768
#define N_    256
#define HID_  200
#define HP_   208      // hidden padded to 13*16
#define NT_   13       // 208 / 16 n-tiles
#define KC1_  24       // 768 / 32 k-chunks (layer 1)
#define KC2_  7        // 224 / 32 k-chunks (layer 2)
#define MT_   4        // j-tile groups per wave -> M = 64 rows per block

typedef __bf16  bf16_t;
typedef bf16_t  v16bf  __attribute__((ext_vector_type(16)));
typedef float   v8f    __attribute__((ext_vector_type(8)));
typedef unsigned int   v8u   __attribute__((ext_vector_type(8)));
typedef unsigned short u16x8 __attribute__((ext_vector_type(8)));

__device__ __forceinline__ unsigned short f2bf(float f) {
  unsigned int u = __builtin_bit_cast(unsigned int, f);
  u += 0x7FFFu + ((u >> 16) & 1u);          // round-to-nearest-even
  return (unsigned short)(u >> 16);
}
__device__ __forceinline__ float bf2f(unsigned short s) {
  return __builtin_bit_cast(float, ((unsigned int)s) << 16);
}

// ---------------- workspace layout (bytes) ----------------
#define OFF_EMBG  0u                                   // B*N*D f32      = 3,145,728
#define OFF_U     (OFF_EMBG + B_*N_*D_*4u)             // B*N*HP f32     =   851,968
#define OFF_V     (OFF_U    + B_*N_*HP_*4u)
#define OFF_PW1   (OFF_V    + B_*N_*HP_*4u)            // 13*24*32*16 bf16 = 319,488
#define OFF_PW2   (OFF_PW1  + NT_*KC1_*32u*16u*2u)     // 13*7*32*16 bf16  =  93,184
#define OFF_S     (OFF_PW2  + NT_*KC2_*32u*16u*2u)     // B*N*N f32      = 1,048,576

// ---------------- 1) gather selected embeddings ----------------
__global__ void gather_kernel(const float* __restrict__ emb,
                              const int* __restrict__ idx,
                              float* __restrict__ embG) {
  const int n = blockIdx.x, b = blockIdx.y, t = threadIdx.x;
  const int src = idx[b * N_ + n];
  const float* s = emb  + ((size_t)b * T_ + src) * D_;
  float*       d = embG + ((size_t)b * N_ + n)   * D_;
  for (int k = t; k < D_; k += 256) d[k] = s[k];
}

// ---------------- 2) U = embG@W1a + b1, V = embG@W1b (padded) ----------------
__global__ void uv_kernel(const float* __restrict__ embG,
                          const float* __restrict__ W1,
                          const float* __restrict__ b1,
                          float* __restrict__ U, float* __restrict__ V) {
  const int bn = blockIdx.x;        // 0 .. B*N-1
  const int h  = threadIdx.x;       // 0 .. 255
  if (h >= HP_) return;
  float u = 0.f, v = 0.f;
  if (h < HID_) {
    const float* e = embG + (size_t)bn * D_;
    for (int d = 0; d < D_; ++d) {
      const float x = e[d];
      u = fmaf(x, W1[(size_t)d * HID_ + h], u);
      v = fmaf(x, W1[(size_t)(D_ + d) * HID_ + h], v);
    }
    u += b1[h];
  }
  U[(size_t)bn * HP_ + h] = u;
  V[(size_t)bn * HP_ + h] = v;
}

// ---------------- 3) pack a weight block into WMMA B-fragment order ----------------
// dst layout: [nt][kc][lane][16] halves; element e of lane: K = kc*32 + (lane>>4)*8 + (e&7) + (e>>3)*16,
// N = nt*16 + (lane&15).  src is row-major [K][HID_].
__global__ void pack_kernel(const float* __restrict__ src, int rowOff, int kValid,
                            int kcCount, unsigned short* __restrict__ dst) {
  const int gid  = blockIdx.x * blockDim.x + threadIdx.x;
  const int lane = gid & 31;
  const int wid  = gid >> 5;
  const int kc   = wid % kcCount;
  const int nt   = wid / kcCount;
  if (nt >= NT_) return;
  const int n  = nt * 16 + (lane & 15);
  const int hi = lane >> 4;
  unsigned short* out = dst + ((size_t)(nt * kcCount + kc) * 32 + lane) * 16;
  #pragma unroll
  for (int e = 0; e < 16; ++e) {
    const int k = kc * 32 + hi * 8 + (e & 7) + ((e >> 3) << 4);
    float f = 0.f;
    if (k < kValid && n < HID_) f = src[(size_t)(rowOff + k) * HID_ + n];
    out[e] = f2bf(f);
  }
}

// ---------------- 4) fused pair MLP: 64 pairs (row i, cols j0..j0+63) per wave ----------
__global__ __launch_bounds__(32) void pair_score_kernel(
    const float* __restrict__ embG, const float* __restrict__ U,
    const float* __restrict__ V, const unsigned short* __restrict__ pW1,
    const unsigned short* __restrict__ pW2, const float* __restrict__ b2,
    const float* __restrict__ W3, const float* __restrict__ b3,
    float* __restrict__ Sws) {
  const int i  = (int)blockIdx.y + 1;          // 1..255
  const int j0 = (int)blockIdx.x * (16 * MT_); // 0,64,128,192
  if (j0 >= i) return;                         // uniform exit: EXEC stays full for WMMA
  const int b    = (int)blockIdx.z;
  const int lane = (int)threadIdx.x;
  const int r    = lane & 15;
  const int hi   = lane >> 4;

  __shared__ __align__(16) float          eI[D_];
  __shared__ __align__(16) unsigned short hbuf[16 * MT_][232]; // h1 (bf16) then h2 (bf16)

  const float* embB = embG + (size_t)b * N_ * D_;
  { // cooperative load of e_i
    const float4* s4 = (const float4*)(embB + (size_t)i * D_);
    float4* d4 = (float4*)eI;
    #pragma unroll
    for (int t = 0; t < 6; ++t) d4[lane + 32 * t] = s4[lane + 32 * t];
  }
  __syncthreads();

  const float* eJ[MT_];
  #pragma unroll
  for (int g = 0; g < MT_; ++g)
    eJ[g] = embB + (size_t)(j0 + g * 16 + r) * D_;   // j <= 255 always in-bounds

  union AFrag { v16bf v; unsigned short s[16]; u16x8 h[2]; };

  // ---- layer 1: (e_i .* e_j) @ W1c, 4 row-groups share every B fragment ----
  v8f acc[MT_][NT_];
  #pragma unroll
  for (int g = 0; g < MT_; ++g)
    #pragma unroll
    for (int nt = 0; nt < NT_; ++nt)
      acc[g][nt] = (v8f){0.f,0.f,0.f,0.f,0.f,0.f,0.f,0.f};

  for (int kc = 0; kc < KC1_; ++kc) {
    const int kb = kc * 32 + hi * 8;
    // e_i chunks shared by all 4 groups
    const float4 a0 = *(const float4*)(eI + kb);
    const float4 a1 = *(const float4*)(eI + kb + 4);
    const float4 a2 = *(const float4*)(eI + kb + 16);
    const float4 a3 = *(const float4*)(eI + kb + 20);
    AFrag A[MT_];
    #pragma unroll
    for (int g = 0; g < MT_; ++g) {
      const float4 c0 = *(const float4*)(eJ[g] + kb);
      const float4 c1 = *(const float4*)(eJ[g] + kb + 4);
      const float4 c2 = *(const float4*)(eJ[g] + kb + 16);
      const float4 c3 = *(const float4*)(eJ[g] + kb + 20);
      A[g].s[0]  = f2bf(a0.x*c0.x); A[g].s[1]  = f2bf(a0.y*c0.y);
      A[g].s[2]  = f2bf(a0.z*c0.z); A[g].s[3]  = f2bf(a0.w*c0.w);
      A[g].s[4]  = f2bf(a1.x*c1.x); A[g].s[5]  = f2bf(a1.y*c1.y);
      A[g].s[6]  = f2bf(a1.z*c1.z); A[g].s[7]  = f2bf(a1.w*c1.w);
      A[g].s[8]  = f2bf(a2.x*c2.x); A[g].s[9]  = f2bf(a2.y*c2.y);
      A[g].s[10] = f2bf(a2.z*c2.z); A[g].s[11] = f2bf(a2.w*c2.w);
      A[g].s[12] = f2bf(a3.x*c3.x); A[g].s[13] = f2bf(a3.y*c3.y);
      A[g].s[14] = f2bf(a3.z*c3.z); A[g].s[15] = f2bf(a3.w*c3.w);
    }
    #pragma unroll
    for (int nt = 0; nt < NT_; ++nt) {
      const v8u raw = *(const v8u*)(pW1 + (((size_t)nt * KC1_ + kc) * 32 + lane) * 16);
      const v16bf Bf = __builtin_bit_cast(v16bf, raw);
      #pragma unroll
      for (int g = 0; g < MT_; ++g)
        acc[g][nt] = __builtin_amdgcn_wmma_f32_16x16x32_bf16(
            false, A[g].v, false, Bf, (short)0, acc[g][nt], false, false);
    }
  }

  // ---- epilogue 1: + U[i] + V[j], ReLU, stash bf16 h1 in LDS ----
  const float* Urow = U + ((size_t)b * N_ + i) * HP_;
  #pragma unroll
  for (int g = 0; g < MT_; ++g) {
    #pragma unroll
    for (int nt = 0; nt < NT_; ++nt) {
      const int n = nt * 16 + r;
      const float u = Urow[n];
      #pragma unroll
      for (int v = 0; v < 8; ++v) {
        const int M = g * 16 + v + 8 * hi;       // C layout: M = vgpr + 8*(lane>=16)
        float val = acc[g][nt][v] + u + V[((size_t)b * N_ + (j0 + M)) * HP_ + n];
        hbuf[M][n] = f2bf(fmaxf(val, 0.f));
      }
    }
  }
  #pragma unroll
  for (int t = 0; t < 32; ++t) {                  // zero K-pad columns 208..223, 64 rows
    const int idx = lane + 32 * t;                // 0..1023
    hbuf[idx >> 4][208 + (idx & 15)] = 0;
  }
  __syncthreads();

  // ---- layer 2: h1 @ W2 (same 4-way B reuse), accumulators reused ----
  #pragma unroll
  for (int g = 0; g < MT_; ++g)
    #pragma unroll
    for (int nt = 0; nt < NT_; ++nt)
      acc[g][nt] = (v8f){0.f,0.f,0.f,0.f,0.f,0.f,0.f,0.f};

  #pragma unroll
  for (int kc = 0; kc < KC2_; ++kc) {
    const int kb = kc * 32 + hi * 8;
    AFrag A2[MT_];
    #pragma unroll
    for (int g = 0; g < MT_; ++g) {
      A2[g].h[0] = *(const u16x8*)&hbuf[g * 16 + r][kb];
      A2[g].h[1] = *(const u16x8*)&hbuf[g * 16 + r][kb + 16];
    }
    #pragma unroll
    for (int nt = 0; nt < NT_; ++nt) {
      const v8u raw = *(const v8u*)(pW2 + (((size_t)nt * KC2_ + kc) * 32 + lane) * 16);
      const v16bf Bf = __builtin_bit_cast(v16bf, raw);
      #pragma unroll
      for (int g = 0; g < MT_; ++g)
        acc[g][nt] = __builtin_amdgcn_wmma_f32_16x16x32_bf16(
            false, A2[g].v, false, Bf, (short)0, acc[g][nt], false, false);
    }
  }
  __syncthreads();

  // ---- epilogue 2: + b2, ReLU -> overwrite hbuf with bf16 h2 ----
  #pragma unroll
  for (int g = 0; g < MT_; ++g) {
    #pragma unroll
    for (int nt = 0; nt < NT_; ++nt) {
      const int n = nt * 16 + r;
      const float bb = (n < HID_) ? b2[n] : 0.f;
      #pragma unroll
      for (int v = 0; v < 8; ++v) {
        const int M = g * 16 + v + 8 * hi;
        hbuf[M][n] = f2bf(fmaxf(acc[g][nt][v] + bb, 0.f));
      }
    }
  }
  __syncthreads();

  // ---- layer 3: per-pair 200-dot with W3 (2 rows per lane) ----
  #pragma unroll
  for (int rr0 = 0; rr0 < 16 * MT_; rr0 += 32) {
    const int rr = rr0 + lane;
    const int j  = j0 + rr;
    if (j < i) {
      float s = b3[0];
      for (int k = 0; k < HID_; ++k) s = fmaf(bf2f(hbuf[rr][k]), W3[k], s);
      Sws[((size_t)b * N_ + i) * N_ + j] = s;
    }
  }
}

// ---------------- 5) masked row softmax (diagonal logit = 0) ----------------
__global__ __launch_bounds__(32) void softmax_kernel(const float* __restrict__ Sws,
                                                     float* __restrict__ out) {
  const int i = blockIdx.x, b = blockIdx.y, lane = threadIdx.x;
  const float* row  = Sws + ((size_t)b * N_ + i) * N_;
  float*       orow = out + ((size_t)b * N_ + i) * N_;
  float m = -3.0e38f;
  for (int j = lane; j <= i; j += 32) m = fmaxf(m, (j == i) ? 0.f : row[j]);
  #pragma unroll
  for (int off = 16; off > 0; off >>= 1) m = fmaxf(m, __shfl_xor(m, off, 32));
  float sum = 0.f;
  for (int j = lane; j <= i; j += 32) sum += __expf(((j == i) ? 0.f : row[j]) - m);
  #pragma unroll
  for (int off = 16; off > 0; off >>= 1) sum += __shfl_xor(sum, off, 32);
  const float inv = 1.f / sum;
  for (int j = lane; j < N_; j += 32) {
    if (j <= i) orow[j] = __expf(((j == i) ? 0.f : row[j]) - m) * inv;
    else        orow[j] = -1000.0f;
  }
}

// ---------------- launch ----------------
extern "C" void kernel_launch(void* const* d_in, const int* in_sizes, int n_in,
                              void* d_out, int out_size, void* d_ws, size_t ws_size,
                              hipStream_t stream) {
  const float* event_embed = (const float*)d_in[0];
  const int*   event_idx   = (const int*)  d_in[1];
  const float* W1 = (const float*)d_in[2];
  const float* b1 = (const float*)d_in[3];
  const float* W2 = (const float*)d_in[4];
  const float* b2 = (const float*)d_in[5];
  const float* W3 = (const float*)d_in[6];
  const float* b3 = (const float*)d_in[7];
  float* out = (float*)d_out;

  char* ws = (char*)d_ws;
  float*          embG = (float*)(ws + OFF_EMBG);
  float*          U    = (float*)(ws + OFF_U);
  float*          V    = (float*)(ws + OFF_V);
  unsigned short* pW1  = (unsigned short*)(ws + OFF_PW1);
  unsigned short* pW2  = (unsigned short*)(ws + OFF_PW2);
  float*          Sws  = (float*)(ws + OFF_S);

  gather_kernel<<<dim3(N_, B_), 256, 0, stream>>>(event_embed, event_idx, embG);
  uv_kernel<<<dim3(B_ * N_), 256, 0, stream>>>(embG, W1, b1, U, V);
  pack_kernel<<<dim3((NT_ * KC1_ * 32) / 256), 256, 0, stream>>>(W1, 2 * D_, D_, KC1_, pW1);
  pack_kernel<<<dim3((NT_ * KC2_ * 32 + 255) / 256), 256, 0, stream>>>(W2, 0, HID_, KC2_, pW2);
  pair_score_kernel<<<dim3(MT_, N_ - 1, B_), 32, 0, stream>>>(
      embG, U, V, pW1, pW2, b2, W3, b3, Sws);
  softmax_kernel<<<dim3(N_, B_), 32, 0, stream>>>(Sws, out);
}
